// GPT2_69209103008395
// MI455X (gfx1250) — matrix operand order repaired and checked by reading, not compiled, not run on hardware
//
#include <hip/hip_runtime.h>
#include <hip/hip_bf16.h>
#include <math.h>

// GPT-2 small forward for MI455X (gfx1250, wave32, WMMA + TDM).
// All GEMMs: f16 x f16 -> f32 accumulate via v_wmma_f32_16x16x32_f16.
// Weights are pre-packed (per launch, serialized on stream) into B-fragment
// f16 tiles; the GEMM streams them into LDS with the Tensor Data Mover
// (tensor_load_to_lds, double-buffered, s_wait_tensorcnt) while WMMAs run.
// Attention: flash-style streaming softmax on f16 QKV; QK^T and P@V via WMMA.

typedef __attribute__((ext_vector_type(16))) _Float16 v16h;
typedef __attribute__((ext_vector_type(8)))  _Float16 v8h;
typedef __attribute__((ext_vector_type(8)))  float    v8f;
typedef __attribute__((ext_vector_type(4)))  unsigned int u32x4;
typedef __attribute__((ext_vector_type(8)))  int      i32x8;
typedef __attribute__((ext_vector_type(4)))  int      i32x4;

#define EE   768
#define E3   2304
#define E4   3072
#define SEQ  1024
#define NB   2
#define NH   12
#define HD   64
#define MM   (NB * SEQ)   // 2048 rows
#define VV   50257
#define VPAD 50432        // round_up(V, 256)

#define WMMA_F16(A_, B_, C_) \
  __builtin_amdgcn_wmma_f32_16x16x32_f16(false, (A_), false, (B_), (short)0, (C_), false, false)

// ---- A-fragment (16x32 f16) loader, this file's contiguous k order (B packed to match) ----
// lane<16 rows hold K {0..15}; lane>=16 hold {16..31}.
// p = per-lane row pointer (row = lane&15) at k0; 16B-aligned.
__device__ __forceinline__ v16h load_a16(const _Float16* p, int lane) {
  const v8h* q = (const v8h*)p;
  v8h lo = q[(lane & 16) ? 2 : 0];
  v8h hi = q[(lane & 16) ? 3 : 1];
  v16h a;
#pragma unroll
  for (int j = 0; j < 8; ++j) { a[j] = lo[j]; a[j + 8] = hi[j]; }
  return a;
}

// ---- TDM: copy 4 packed B tiles (4 x 1KB, stride kt32*1KB) into LDS ----
// D# per ISA 8.3/8.4: 2D tile, data_size=2B, tile_dim0=512, tile_dim1=4.
// This toolchain exposes the 6-arg tensor_load_to_lds (extra i32x8 group).
__device__ __forceinline__ void tdm_load_b(const _Float16* gbase, unsigned lds_off,
                                           int kt32) {
  unsigned long long ga = (unsigned long long)(size_t)gbase;
  u32x4 g0;
  g0[0] = 1u;                                  // count=1 (user D#, valid)
  g0[1] = lds_off;                             // LDS byte address
  g0[2] = (unsigned)ga;                        // global_addr[31:0]
  g0[3] = (unsigned)(ga >> 32) | (2u << 30);   // global_addr[56:32] | type=2
  i32x8 g1;
  g1[0] = 0x00010000;                          // wg_mask=0, data_size=2B
  g1[1] = 512 << 16;                           // tensor_dim0 = 512 (lo16)
  g1[2] = 4 << 16;                             // tensor_dim0 hi | tensor_dim1=4
  g1[3] = 512 << 16;                           // tensor_dim1 hi | tile_dim0=512
  g1[4] = 4;                                   // tile_dim1=4, tile_dim2=0
  g1[5] = kt32 * 512;                          // tensor_dim0_stride (elements)
  g1[6] = 0;                                   // stride hi | dim1_stride lo
  g1[7] = 0;
  i32x4 gz4 = {0, 0, 0, 0};                    // groups 2/3 unused (2D tile)
  i32x8 gz8 = {0, 0, 0, 0, 0, 0, 0, 0};
  __builtin_amdgcn_tensor_load_to_lds(g0, g1, gz4, gz4, gz8, 0);
}

// ---------------- embedding ----------------
__global__ __launch_bounds__(256) void embed_kernel(const int* __restrict__ tok,
                                                    const float* __restrict__ wte,
                                                    const float* __restrict__ wpe,
                                                    float* __restrict__ x) {
  int row = blockIdx.x;               // b*S + s
  int s = row & (SEQ - 1);
  int t = tok[row];
  const float* we = wte + (size_t)t * EE;
  const float* wp = wpe + (size_t)s * EE;
  float* xr = x + (size_t)row * EE;
  for (int e = threadIdx.x; e < EE; e += 256) xr[e] = we[e] + wp[e];
}

// ---------------- layernorm (f32 in -> f16 out) ----------------
__global__ __launch_bounds__(256) void layernorm_kernel(const float* __restrict__ x,
                                                        const float* __restrict__ g,
                                                        const float* __restrict__ beta,
                                                        _Float16* __restrict__ outh) {
  __shared__ float red[256];
  int row = blockIdx.x, tid = threadIdx.x;
  const float* xr = x + (size_t)row * EE;
  float v0 = xr[tid], v1 = xr[tid + 256], v2 = xr[tid + 512];
  red[tid] = v0 + v1 + v2;
  __syncthreads();
#pragma unroll
  for (int o = 128; o > 0; o >>= 1) {
    if (tid < o) red[tid] += red[tid + o];
    __syncthreads();
  }
  float mu = red[0] * (1.0f / 768.0f);
  __syncthreads();
  float d0 = v0 - mu, d1 = v1 - mu, d2 = v2 - mu;
  red[tid] = d0 * d0 + d1 * d1 + d2 * d2;
  __syncthreads();
#pragma unroll
  for (int o = 128; o > 0; o >>= 1) {
    if (tid < o) red[tid] += red[tid + o];
    __syncthreads();
  }
  float rstd = rsqrtf(red[0] * (1.0f / 768.0f) + 1e-5f);
  _Float16* orow = outh + (size_t)row * EE;
  orow[tid]       = (_Float16)(d0 * rstd * g[tid]       + beta[tid]);
  orow[tid + 256] = (_Float16)(d1 * rstd * g[tid + 256] + beta[tid + 256]);
  orow[tid + 512] = (_Float16)(d2 * rstd * g[tid + 512] + beta[tid + 512]);
}

// ---------------- weight pack: f32 [K,N] -> f16 B-fragment tiles ----------------
// Tile (nt,kt) covers cols nt*16..+15, K rows kt*32..+31, stored as 512 halves:
// half[lane*16 + j] = W[kt*32 + (lane&16?16:0) + j][nt*16 + (lane&15)], zero pad.
__global__ __launch_bounds__(256) void pack_w_kernel(const float* __restrict__ W,
                                                     _Float16* __restrict__ P,
                                                     int K, int N, int tiles) {
  int lane = threadIdx.x & 31;
  int wave = threadIdx.x >> 5;
  int w = blockIdx.x * 8 + wave;
  if (w >= tiles) return;
  int kt32 = K >> 5;
  int nt = w / kt32;
  int kt = w - nt * kt32;
  int col = nt * 16 + (lane & 15);
  bool ok = col < N;
  const float* wp = W + (size_t)(kt * 32 + ((lane & 16) ? 16 : 0)) * N + col;
  v16h o;
#pragma unroll
  for (int j = 0; j < 16; ++j)
    o[j] = ok ? (_Float16)wp[(size_t)j * N] : (_Float16)0.0f;
  *(v16h*)(P + (size_t)w * 512 + lane * 16) = o;
}

// ---------------- V transpose: qkv f16 [M,3E] -> vt [B,NH,HD,SEQ] ----------------
__global__ __launch_bounds__(256) void vtrans_kernel(const _Float16* __restrict__ qkvh,
                                                     _Float16* __restrict__ vt) {
  int row = blockIdx.x;               // b*S + s
  int b = row >> 10, s = row & (SEQ - 1);
  const _Float16* vr = qkvh + (size_t)row * E3 + 2 * EE;
  for (int e = threadIdx.x; e < EE; e += 256) {
    int h = e >> 6, d = e & 63;
    vt[(size_t)((b * NH + h) * HD + d) * SEQ + s] = vr[e];
  }
}

// ---------------- WMMA GEMM: out = act(A@W + bias) [+ residual] ----------------
// A: [M,K] f16 row-major. Pw: packed f16 B tiles (Npad cols, zero-padded).
// Wave tile 32x64 (8 wmma/k-step), block tile 64x256 (8 waves, 2x4).
// B tiles stream global->LDS via TDM, double-buffered on TENSORcnt.
template <bool GELU, bool OUT_F16>
__global__ __launch_bounds__(256) void gemm_kernel(const _Float16* __restrict__ A,
                                                   const _Float16* __restrict__ Pw,
                                                   const float* __restrict__ bias,
                                                   const float* __restrict__ residual,
                                                   void* __restrict__ out,
                                                   int M, int N, int K) {
  __shared__ __align__(16) _Float16 lds_b[8][2][4 * 512];   // 64KB / workgroup
  int lane = threadIdx.x & 31;
  int wave = threadIdx.x >> 5;
  int m0 = blockIdx.y * 64 + (wave >> 2) * 32;
  int n0 = blockIdx.x * 256 + (wave & 3) * 64;
  int kt32 = K >> 5;
  int nt0 = n0 >> 4;
  if (m0 >= M) return;

  const _Float16* arow0 = A + (size_t)(m0 + (lane & 15)) * K;
  const _Float16* arow1 = arow0 + (size_t)16 * K;
  const _Float16* btile0 = Pw + (size_t)nt0 * kt32 * 512;   // +kt*512 per step
  unsigned lds_off[2];
  lds_off[0] = (unsigned)(size_t)&lds_b[wave][0][0];        // LDS aperture: low
  lds_off[1] = (unsigned)(size_t)&lds_b[wave][1][0];        // 32 bits = offset

  tdm_load_b(btile0, lds_off[0], kt32);                     // prologue fill
  v8f acc[2][4] = {};
  int db = 0;
  for (int kt = 0; kt < kt32; ++kt) {
    if (kt + 1 < kt32) {
      tdm_load_b(btile0 + (size_t)(kt + 1) * 512, lds_off[db ^ 1], kt32);
      __builtin_amdgcn_s_wait_tensorcnt(1);  // in-order TDM: older buf ready
    } else {
      __builtin_amdgcn_s_wait_tensorcnt(0);
    }
    asm volatile("" ::: "memory");           // LDS was written by TDM

    int k0 = kt * 32;
    v16h a0 = load_a16(arow0 + k0, lane);
    v16h a1 = load_a16(arow1 + k0, lane);
    v16h b[4];
#pragma unroll
    for (int c = 0; c < 4; ++c)
      b[c] = *(const v16h*)&lds_b[wave][db][c * 512 + lane * 16];
#pragma unroll
    for (int c = 0; c < 4; ++c) {
      acc[0][c] = WMMA_F16(a0, b[c], acc[0][c]);
      acc[1][c] = WMMA_F16(a1, b[c], acc[1][c]);
    }
    db ^= 1;
  }

  int off = (lane & 16) ? 8 : 0;
  int nc = lane & 15;
#pragma unroll
  for (int i = 0; i < 2; ++i)
#pragma unroll
    for (int c = 0; c < 4; ++c)
#pragma unroll
      for (int r = 0; r < 8; ++r) {
        int m = m0 + i * 16 + off + r;
        int n = n0 + c * 16 + nc;
        if (n < N) {
          float v = acc[i][c][r];
          if (bias) v += bias[n];
          if (GELU) v = 0.5f * v * (1.0f + erff(v * 0.70710678118f));
          if (residual) v += residual[(size_t)m * N + n];
          if (OUT_F16)
            ((_Float16*)out)[(size_t)m * N + n] = (_Float16)v;
          else
            ((float*)out)[(size_t)m * N + n] = v;
        }
      }
}

// ---------------- flash attention (one wave = 16 queries x one head) ----------------
__global__ __launch_bounds__(256) void attn_kernel(const _Float16* __restrict__ qkvh,
                                                   const _Float16* __restrict__ vt,
                                                   _Float16* __restrict__ oA) {
  __shared__ __align__(32) _Float16 lds_p[8][16][32];
  int lane = threadIdx.x & 31;
  int wave = threadIdx.x >> 5;
  int w = blockIdx.x * 8 + wave;        // 0..1535
  int b = w / (NH * (SEQ / 16));
  int rem = w % (NH * (SEQ / 16));
  int h = rem / (SEQ / 16);
  int qt = rem % (SEQ / 16);
  int off = (lane & 16) ? 8 : 0;
  int nc = lane & 15;
  int sel16 = (lane & 16) ? 16 : 0;
  int qmax = qt * 16 + 15;

  // Q fragments (16x64 -> two 16x32 chunks), straight f16 loads
  v16h qa[2];
#pragma unroll
  for (int kc = 0; kc < 2; ++kc)
    qa[kc] = load_a16(qkvh + (size_t)(b * SEQ + qt * 16 + nc) * E3 + h * HD + kc * 32, lane);

  float m_i[8], l_i[8];
  v8f oacc[4] = {};
#pragma unroll
  for (int r = 0; r < 8; ++r) { m_i[r] = -3.0e38f; l_i[r] = 0.0f; }

  int nkb = (16 * qt + 47) >> 5;        // #32-key blocks covering keys 0..qmax
  for (int kb = 0; kb < nkb; ++kb) {
    int kb32 = kb * 32;
    v8f st[2];
    bool valid1 = (kb32 + 16) <= qmax;
#pragma unroll
    for (int t = 0; t < 2; ++t) {
      if (t == 1 && !valid1) {
#pragma unroll
        for (int r = 0; r < 8; ++r) st[1][r] = -3.0e38f;
        continue;
      }
      int key = kb32 + t * 16 + nc;     // this lane's column (key index)
      v8f s = {};
#pragma unroll
      for (int kc = 0; kc < 2; ++kc) {
        // B-frag of K^T: contiguous 16 features for this lane's key
        v16h kf = *(const v16h*)(qkvh + (size_t)(b * SEQ + key) * E3 + EE + h * HD +
                                 kc * 32 + sel16);
        s = WMMA_F16(qa[kc], kf, s);
      }
#pragma unroll
      for (int r = 0; r < 8; ++r) {
        int q = qt * 16 + off + r;
        s[r] = (key > q) ? -3.0e38f : s[r] * 0.125f;   // causal mask + 1/sqrt(D)
      }
      st[t] = s;
    }

    // online softmax update (row stats shared across each 16-lane half)
#pragma unroll
    for (int r = 0; r < 8; ++r) {
      float rm = fmaxf(st[0][r], st[1][r]);
#pragma unroll
      for (int msk = 1; msk < 16; msk <<= 1)
        rm = fmaxf(rm, __shfl_xor(rm, msk, 32));
      float newm = fmaxf(m_i[r], rm);
      float alpha = __expf(m_i[r] - newm);
      float p0 = __expf(st[0][r] - newm);
      float p1 = __expf(st[1][r] - newm);
      float rs = p0 + p1;
#pragma unroll
      for (int msk = 1; msk < 16; msk <<= 1)
        rs += __shfl_xor(rs, msk, 32);
      l_i[r] = l_i[r] * alpha + rs;
      m_i[r] = newm;
      st[0][r] = p0;
      st[1][r] = p1;
#pragma unroll
      for (int dd = 0; dd < 4; ++dd) oacc[dd][r] *= alpha;
    }

    // P: C-layout -> LDS (row-major 16x32) -> A-fragment layout.
    // Same-wave DS ops are in-order on CDNA5; no workgroup barrier needed.
#pragma unroll
    for (int t = 0; t < 2; ++t)
#pragma unroll
      for (int r = 0; r < 8; ++r)
        lds_p[wave][off + r][t * 16 + nc] = (_Float16)st[t][r];
    asm volatile("" ::: "memory");
    v16h pa = load_a16(&lds_p[wave][nc][0], lane);

    // O += P @ V over this 32-key block; V^T gives contiguous key runs
#pragma unroll
    for (int dd = 0; dd < 4; ++dd) {
      v16h vf = *(const v16h*)(vt + (size_t)((b * NH + h) * HD + dd * 16 + nc) * SEQ +
                               kb32 + sel16);
      oacc[dd] = WMMA_F16(pa, vf, oacc[dd]);
    }
  }

  // normalize + store f16 output ([B,S,E], head h at cols h*64..)
#pragma unroll
  for (int dd = 0; dd < 4; ++dd)
#pragma unroll
    for (int r = 0; r < 8; ++r) {
      int q = qt * 16 + off + r;
      oA[(size_t)(b * SEQ + q) * EE + h * HD + dd * 16 + nc] =
          (_Float16)(oacc[dd][r] / l_i[r]);
    }
}

// ---------------- host-side launch ----------------
static inline int pack_tiles(int K, int Npad) { return (Npad / 16) * (K / 32); }

extern "C" void kernel_launch(void* const* d_in, const int* in_sizes, int n_in,
                              void* d_out, int out_size, void* d_ws, size_t ws_size,
                              hipStream_t stream) {
  const int*   tok    = (const int*)d_in[0];
  const float* wte    = (const float*)d_in[1];
  const float* wpe    = (const float*)d_in[2];
  const float* ln1_g  = (const float*)d_in[3];
  const float* ln1_b  = (const float*)d_in[4];
  const float* attn_w = (const float*)d_in[5];
  const float* attn_b = (const float*)d_in[6];
  const float* proj_w = (const float*)d_in[7];
  const float* proj_b = (const float*)d_in[8];
  const float* ln2_g  = (const float*)d_in[9];
  const float* ln2_b  = (const float*)d_in[10];
  const float* fc_w   = (const float*)d_in[11];
  const float* fc_b   = (const float*)d_in[12];
  const float* fc2_w  = (const float*)d_in[13];
  const float* fc2_b  = (const float*)d_in[14];
  const float* lnf_g  = (const float*)d_in[15];
  const float* lnf_b  = (const float*)d_in[16];
  const float* lm_w   = (const float*)d_in[17];
  float* logits = (float*)d_out;

  char* ws = (char*)d_ws;
  float*    x    = (float*)ws;    ws += (size_t)MM * EE * 4;   // residual stream
  _Float16* hA   = (_Float16*)ws; ws += (size_t)MM * EE * 2;   // LN output (f16 A)
  _Float16* hB   = (_Float16*)ws; ws += (size_t)MM * E4 * 2;   // GELU(fc) output
  _Float16* qkvh = (_Float16*)ws; ws += (size_t)MM * E3 * 2;   // fused QKV (f16)
  _Float16* vt   = (_Float16*)ws; ws += (size_t)NB * NH * HD * SEQ * 2; // V^T
  _Float16* oA   = (_Float16*)ws; ws += (size_t)MM * EE * 2;   // attention out
  _Float16* pw   = (_Float16*)ws;                              // packed weights
                                                               // (max 768*50432*2B)
  dim3 blk(256);
  embed_kernel<<<MM, blk, 0, stream>>>(tok, wte, wpe, x);

  for (int i = 0; i < 4; ++i) {
    // QKV projection
    layernorm_kernel<<<MM, blk, 0, stream>>>(x, ln1_g + i * EE, ln1_b + i * EE, hA);
    int t1 = pack_tiles(EE, E3);
    pack_w_kernel<<<(t1 + 7) / 8, blk, 0, stream>>>(attn_w + (size_t)i * EE * E3, pw, EE, E3, t1);
    gemm_kernel<false, true><<<dim3(E3 / 256, MM / 64), blk, 0, stream>>>(
        hA, pw, attn_b + (size_t)i * E3, nullptr, qkvh, MM, E3, EE);

    // attention
    vtrans_kernel<<<MM, blk, 0, stream>>>(qkvh, vt);
    attn_kernel<<<(NB * NH * (SEQ / 16)) / 8, blk, 0, stream>>>(qkvh, vt, oA);

    // output projection (+ residual)
    int t2 = pack_tiles(EE, EE);
    pack_w_kernel<<<(t2 + 7) / 8, blk, 0, stream>>>(proj_w + (size_t)i * EE * EE, pw, EE, EE, t2);
    gemm_kernel<false, false><<<dim3(EE / 256, MM / 64), blk, 0, stream>>>(
        oA, pw, proj_b + (size_t)i * EE, x, x, MM, EE, EE);

    // MLP
    layernorm_kernel<<<MM, blk, 0, stream>>>(x, ln2_g + i * EE, ln2_b + i * EE, hA);
    int t3 = pack_tiles(EE, E4);
    pack_w_kernel<<<(t3 + 7) / 8, blk, 0, stream>>>(fc_w + (size_t)i * EE * E4, pw, EE, E4, t3);
    gemm_kernel<true, true><<<dim3(E4 / 256, MM / 64), blk, 0, stream>>>(
        hA, pw, fc_b + (size_t)i * E4, nullptr, hB, MM, E4, EE);
    int t4 = pack_tiles(E4, EE);
    pack_w_kernel<<<(t4 + 7) / 8, blk, 0, stream>>>(fc2_w + (size_t)i * E4 * EE, pw, E4, EE, t4);
    gemm_kernel<false, false><<<dim3(EE / 256, MM / 64), blk, 0, stream>>>(
        hB, pw, fc2_b + (size_t)i * EE, x, x, MM, EE, E4);
  }

  // final LN + lm head
  layernorm_kernel<<<MM, blk, 0, stream>>>(x, lnf_g, lnf_b, hA);
  int t5 = pack_tiles(EE, VPAD);
  pack_w_kernel<<<(t5 + 7) / 8, blk, 0, stream>>>(lm_w, pw, EE, VV, t5);
  gemm_kernel<false, false><<<dim3(VPAD / 256, MM / 64), blk, 0, stream>>>(
      hA, pw, nullptr, nullptr, logits, MM, VV, EE);
}
